// _Capsule_5841155523250
// MI455X (gfx1250) — compile-verified, matrix operand order checked by reading, not run on hardware
//
#include <hip/hip_runtime.h>

typedef __attribute__((ext_vector_type(16))) _Float16 v16h;
typedef __attribute__((ext_vector_type(8)))  float    v8f;

#define IN_CAPS 1152
#define OUT_CH  32
#define NJ      9     // row-iterations: 32 waves * 4 rows * 9 = 1152

// ---- all-VALU cross-lane helpers (wave32) ----
template <int CTRL>
__device__ __forceinline__ float dpp_xadd(float x) {
    // x + value from lane (lane ^ m) via DPP16; all lanes valid
    int y = __builtin_amdgcn_update_dpp(0, __float_as_int(x), CTRL, 0xF, 0xF, true);
    return x + __int_as_float(y);
}
__device__ __forceinline__ float xor16_add(float x) {
    // swap 16-lane halves with identity select, then add
    int xi = __float_as_int(x);
    int y  = __builtin_amdgcn_permlanex16(xi, xi, 0x76543210, (int)0xFEDCBA98, false, false);
    return x + __int_as_float(y);
}
__device__ __forceinline__ float lane_sum8(float x) {   // sum within 8-lane groups
    x = dpp_xadd<0xB1>(x);    // quad_perm [1,0,3,2]  : xor 1
    x = dpp_xadd<0x4E>(x);    // quad_perm [2,3,0,1]  : xor 2
    x = dpp_xadd<0x164>(x);   // ROW_XMASK 4          : xor 4
    return x;
}
__device__ __forceinline__ float lane_sum32(float x) {  // sum across all 32 lanes
    x = lane_sum8(x);
    x = dpp_xadd<0x168>(x);   // ROW_XMASK 8          : xor 8
    x = xor16_add(x);         // xor 16
    return x;
}

// Cross-wave column reduction: total[k] = sum_w part[w][k], done by wave 0.
// D = ones(16x32) * P(32x16) with v_wmma_f32_16x16x32_f16, two tiles for
// channels 0-15 / 16-31. All rows of D are identical (A all ones), so d[0]
// in lane l is colsum of channel l%16 of that tile; picking tile by (l<16)
// yields colsum[l] in lane l.
__device__ __forceinline__ float wmma_colsum(const float part[32][32], int l) {
    v16h a;
#pragma unroll
    for (int i = 0; i < 16; ++i) a[i] = (_Float16)1.0f;

    const int rbase = (l & 16);   // lanes 16-31 supply B rows K=16..31
    const int col0  = (l & 15);
    float res0 = 0.0f, res1 = 0.0f;
#pragma unroll
    for (int h = 0; h < 2; ++h) {
        const int col = col0 + 16 * h;
        v16h b;
#pragma unroll
        for (int j = 0; j < 8; ++j) {   // B VGPR j: rows (2j,2j+1) [+16 for hi lanes]
            b[2 * j]     = (_Float16)part[rbase + 2 * j][col];
            b[2 * j + 1] = (_Float16)part[rbase + 2 * j + 1][col];
        }
        v8f c = {};
        v8f d = __builtin_amdgcn_wmma_f32_16x16x32_f16(
            false, a, false, b, (short)0, c, false, false);
        if (h == 0) res0 = d[0]; else res1 = d[0];
    }
    return (l < 16) ? res0 : res1;
}

__device__ __forceinline__ float squash_lane(float s) {
    // v = (n2/(1+n2)) * s / sqrt(n2+eps), n2 summed across 32 lanes
    float n2  = lane_sum32(s * s);
    float den = (1.0f + n2) * __builtin_amdgcn_sqrtf(n2 + 1e-8f);
    return s * n2 * __builtin_amdgcn_rcpf(den);
}

__global__ __launch_bounds__(1024)
void caps_route_kernel(const float* __restrict__ U,   // (N, 1152, 32)
                       const float* __restrict__ C0,  // (1, 1152, 32)
                       float* __restrict__ out)       // (N, 1, 32)
{
    const int n   = blockIdx.x;
    const int tid = threadIdx.x;
    const int w   = tid >> 5;          // wave 0..31
    const int l   = tid & 31;
    const int rq  = l >> 3;            // row-in-quad 0..3
    const int cg  = l & 7;             // channel group: channels 4*cg .. 4*cg+3

    __shared__ float part[32][32];     // per-wave partial column sums
    __shared__ float vsh[32];          // broadcast of v

    // row(j) = 4*w + rq + 128*j ; float4 index into row-major (1152,32)
    const float4* U4 = (const float4*)(U + (size_t)n * (IN_CAPS * OUT_CH));
    const float4* C4 = (const float4*)C0;
    const int base4  = (4 * w + rq) * (OUT_CH / 4) + cg;   // float4 units
    const int step4  = 128 * (OUT_CH / 4);                 // 128 rows per j

    // Load this thread's 9 float4 tiles of u (contiguous 512B per wave per j)
    float4 u[NJ];
#pragma unroll
    for (int j = 0; j < NJ; ++j) u[j] = U4[base4 + j * step4];

    // ---------- pass 1: s = sum_i u*c0 ----------
    float4 acc = make_float4(0.f, 0.f, 0.f, 0.f);
#pragma unroll
    for (int j = 0; j < NJ; ++j) {
        float4 c = C4[base4 + j * step4];
        acc.x = fmaf(u[j].x, c.x, acc.x);
        acc.y = fmaf(u[j].y, c.y, acc.y);
        acc.z = fmaf(u[j].z, c.z, acc.z);
        acc.w = fmaf(u[j].w, c.w, acc.w);
    }

    float4 vk;
    for (int r = 0; r < 3; ++r) {
        // fold the 4 row-groups: lanes differing in bit3 (xor8) and bit4 (xor16)
        acc.x = xor16_add(dpp_xadd<0x168>(acc.x));
        acc.y = xor16_add(dpp_xadd<0x168>(acc.y));
        acc.z = xor16_add(dpp_xadd<0x168>(acc.z));
        acc.w = xor16_add(dpp_xadd<0x168>(acc.w));
        if (rq == 0) *(float4*)&part[w][4 * cg] = acc;   // lanes 0-7 hold full wave sums
        __syncthreads();

        if (w == 0) {
            float s = wmma_colsum(part, l);   // colsum[l] via 2 WMMAs
            float v = squash_lane(s);
            if (r == 2) out[(size_t)n * OUT_CH + l] = v * 0.5f + 0.5f;
            vsh[l] = v;
        }
        __syncthreads();
        if (r == 2) break;
        vk = *(const float4*)&vsh[4 * cg];    // this lane's 4 channels of v
        __syncthreads();                      // vsh/part reuse protection

        // ---------- routing pass: c = softmax_k(u*v); s = sum_i u*c ----------
        acc = make_float4(0.f, 0.f, 0.f, 0.f);
#pragma unroll
        for (int j = 0; j < NJ; ++j) {
            const float LOG2E = 1.44269504088896f;
            float e0 = __builtin_amdgcn_exp2f(u[j].x * vk.x * LOG2E);
            float e1 = __builtin_amdgcn_exp2f(u[j].y * vk.y * LOG2E);
            float e2 = __builtin_amdgcn_exp2f(u[j].z * vk.z * LOG2E);
            float e3 = __builtin_amdgcn_exp2f(u[j].w * vk.w * LOG2E);
            // row sum over 32 channels: 3 in-lane adds + 3-step butterfly over 8 lanes
            float rs = lane_sum8((e0 + e1) + (e2 + e3));
            float ri = __builtin_amdgcn_rcpf(rs);
            acc.x = fmaf(u[j].x * e0, ri, acc.x);   // += u * softmax(u*v)
            acc.y = fmaf(u[j].y * e1, ri, acc.y);
            acc.z = fmaf(u[j].z * e2, ri, acc.z);
            acc.w = fmaf(u[j].w * e3, ri, acc.w);
        }
    }
}

extern "C" void kernel_launch(void* const* d_in, const int* in_sizes, int n_in,
                              void* d_out, int out_size, void* d_ws, size_t ws_size,
                              hipStream_t stream) {
    (void)n_in; (void)d_ws; (void)ws_size; (void)out_size;
    const float* U  = (const float*)d_in[0];   // u_hat (N,1152,32) f32
    const float* C0 = (const float*)d_in[1];   // c     (1,1152,32) f32
    float* out      = (float*)d_out;           // (N,1,32) f32

    const int n_samples = in_sizes[0] / (IN_CAPS * OUT_CH);  // 4096
    dim3 grid(n_samples), block(1024);
    caps_route_kernel<<<grid, block, 0, stream>>>(U, C0, out);
}
